// GCN_88218628260128
// MI455X (gfx1250) — matrix operand order repaired and checked by reading, not compiled, not run on hardware
//
#include <hip/hip_runtime.h>
#include <hip/hip_bf16.h>

typedef __attribute__((ext_vector_type(16))) _Float16 v16h;
typedef __attribute__((ext_vector_type(8)))  float    v8f;

#define F_DIM   256
#define EMBJ    63
#define HID     128
#define OUT_DIM 7

// ---------------------------------------------------------------------------
// Stats: per-column sum / sumsq (population std), block-partial + f32 atomics
// ---------------------------------------------------------------------------
__global__ void k_init(float* colsum, float* colss, float* cvec, float* deg, int n) {
    int g = blockIdx.x * blockDim.x + threadIdx.x;
    if (g < F_DIM) { colsum[g] = 0.f; colss[g] = 0.f; }
    if (g < HID)   cvec[g] = 0.f;
    if (g < n)     deg[g] = 1.0f;          // self-loop contribution
}

__global__ void k_colstats(const float* __restrict__ x, float* colsum, float* colss, int n) {
    int col = threadIdx.x;                 // blockDim.x == 256
    int rpb = (n + gridDim.x - 1) / gridDim.x;
    int r0 = blockIdx.x * rpb;
    int r1 = min(r0 + rpb, n);
    float s = 0.f, ss = 0.f;
    for (int r = r0; r < r1; ++r) {
        float v = x[(size_t)r * F_DIM + col];   // coalesced across the block
        s += v; ss += v * v;
    }
    atomicAdd(&colsum[col], s);
    atomicAdd(&colss[col], ss);
}

__global__ void k_finalize_stats(const float* colsum, const float* colss,
                                 float* meanp, float* istdp, int n) {
    int c = threadIdx.x;                   // 256 threads
    float m   = colsum[c] / (float)n;
    float var = colss[c] / (float)n - m * m;
    float sd  = sqrtf(fmaxf(var, 0.f));
    if (sd == 0.f) sd = 1.f;
    meanp[c] = m;
    istdp[c] = 1.f / sd;
}

// ---------------------------------------------------------------------------
// Degree / symmetric norm (with self loops)
// ---------------------------------------------------------------------------
__global__ void k_deg(const int* __restrict__ ei, float* deg, int e) {
    int g = blockIdx.x * blockDim.x + threadIdx.x;
    if (g < e) atomicAdd(&deg[ei[e + g]], 1.0f);   // dst row of edge_index
}

__global__ void k_dinv(float* d, int n) {
    int g = blockIdx.x * blockDim.x + threadIdx.x;
    if (g < n) { float v = d[g]; d[g] = (v > 0.f) ? rsqrtf(v) : 0.f; }
}

// ---------------------------------------------------------------------------
// Fold the node-independent embedding columns of feats@W1 into a constant
// C[h] = sum_f sum_{j<63} emb[f,j] * W1[f*64+j, h]
// ---------------------------------------------------------------------------
__global__ void k_fold_emb(const float* __restrict__ emb, const float* __restrict__ W1,
                           float* cvec) {
    int f = blockIdx.x;                    // 256 blocks
    int h = threadIdx.x;                   // 128 threads, W1 reads coalesced
    const float* wrow = W1 + (size_t)f * 64 * HID;
    float acc = 0.f;
    for (int j = 0; j < EMBJ; ++j)
        acc += emb[f * EMBJ + j] * wrow[(size_t)j * HID + h];
    atomicAdd(&cvec[h], acc);
}

// W1v^T (f16, N-major [128][256]): B fragments become contiguous 32B loads
__global__ void k_prep_w1vT(const float* __restrict__ W1, _Float16* w1vT) {
    int f = blockIdx.x, h = threadIdx.x;   // 256 x 128
    w1vT[(size_t)h * F_DIM + f] = (_Float16)W1[((size_t)f * 64 + 63) * HID + h];
}

// W2^T padded to 16 columns (f16, [16][128]); cols 7..15 are zero
__global__ void k_prep_w2T(const float* __restrict__ W2, _Float16* w2T) {
    int nn = blockIdx.x, k = threadIdx.x;  // 16 x 128
    w2T[nn * HID + k] = (nn < OUT_DIM) ? (_Float16)W2[(size_t)k * OUT_DIM + nn]
                                       : (_Float16)0.f;
}

// ---------------------------------------------------------------------------
// GEMM1: hpre[N,128] = normalize(x)[N,256] @ W1v[256,128] + C
// One wave per 16x16 output tile; 8 x v_wmma_f32_16x16x32_f16 per tile.
// ---------------------------------------------------------------------------
__global__ __launch_bounds__(32) void k_gemm1_wmma(
    const float* __restrict__ x, const float* __restrict__ meanp,
    const float* __restrict__ istdp, const _Float16* __restrict__ w1vT,
    const float* __restrict__ cvec, float* __restrict__ hpre, int n)
{
    const int lane = threadIdx.x & 31;
    const int half = lane >> 4;            // 0: lanes 0-15, 1: lanes 16-31
    const int l16  = lane & 15;
    const int row0 = blockIdx.x * 16;
    const int col0 = blockIdx.y * 16;
    int ar = row0 + l16; if (ar >= n) ar = n - 1;     // clamp (EXEC must stay all-1)
    const float*    xrow = x    + (size_t)ar * F_DIM;
    const _Float16* bcol = w1vT + (size_t)(col0 + l16) * F_DIM;

    v8f acc = {};
#pragma unroll 2
    for (int k0 = 0; k0 < F_DIM; k0 += 32) {
        const int ka = k0 + half * 16;     // A: contiguous [ka..ka+15]
        const int kb = k0 + half * 16;     // B: contiguous [kb..kb+15]
        v16h a;
#pragma unroll
        for (int i = 0; i < 8; ++i) {
            a[i]     = (_Float16)((xrow[ka + i]      - meanp[ka + i])      * istdp[ka + i]);
            a[i + 8] = (_Float16)((xrow[ka + 8 + i] - meanp[ka + 8 + i]) * istdp[ka + 8 + i]);
        }
        v16h b = *(const v16h*)(bcol + kb);
        acc = __builtin_amdgcn_wmma_f32_16x16x32_f16(false, a, false, b,
                                                     (short)0, acc, false, false);
    }
    // D layout: VGPR r -> M = m0+r, N = l16. Hoist bound check to one branch:
    const int   m0   = row0 + half * 8;
    const float cadd = cvec[col0 + l16];
    float* outp = hpre + (size_t)m0 * HID + col0 + l16;
    if (row0 + 16 <= n) {                  // full tile (always, when N%16==0)
#pragma unroll
        for (int r = 0; r < 8; ++r)
            outp[(size_t)r * HID] = acc[r] + cadd;
    } else {
#pragma unroll
        for (int r = 0; r < 8; ++r)
            if (m0 + r < n) outp[(size_t)r * HID] = acc[r] + cadd;
    }
}

// ---------------------------------------------------------------------------
// GCN aggregation 1: agg1 = scatter_add(norm * hpre[src]) + selfloop + b1
// ---------------------------------------------------------------------------
__global__ void k_agg1_init(const float* __restrict__ hpre, const float* __restrict__ dinv,
                            const float* __restrict__ b1, float* __restrict__ agg1, int n) {
    int g = blockIdx.x * blockDim.x + threadIdx.x;
    if (g >= n * HID) return;
    int nn = g >> 7, h = g & 127;
    float di = dinv[nn];
    agg1[g] = hpre[g] * di * di + b1[h];
}

__global__ void k_agg1_edges(const int* __restrict__ ei, const float* __restrict__ hpre,
                             const float* __restrict__ dinv, float* __restrict__ agg1, int e) {
    int g = blockIdx.x * blockDim.x + threadIdx.x;   // e*32 threads, 4 channels each
    if (g >= e * 32) return;
    int edge = g >> 5;
    int q    = (g & 31) << 2;
    int src = ei[edge], dst = ei[e + edge];
    float norm = dinv[src] * dinv[dst];
    const float4 v = *(const float4*)(hpre + (size_t)src * HID + q);
    float* outp = agg1 + (size_t)dst * HID + q;
    atomicAdd(outp + 0, v.x * norm);
    atomicAdd(outp + 1, v.y * norm);
    atomicAdd(outp + 2, v.z * norm);
    atomicAdd(outp + 3, v.w * norm);
}

// ---------------------------------------------------------------------------
// GEMM2 (fused ReLU): z[N,7] = relu(agg1)[N,128] @ W2[128,7]  (N padded to 16)
// ---------------------------------------------------------------------------
__global__ __launch_bounds__(32) void k_gemm2_wmma(
    const float* __restrict__ agg1, const _Float16* __restrict__ w2T,
    float* __restrict__ z, int n)
{
    const int lane = threadIdx.x & 31;
    const int half = lane >> 4;
    const int l16  = lane & 15;
    const int row0 = blockIdx.x * 16;
    int ar = row0 + l16; if (ar >= n) ar = n - 1;
    const float*    arow = agg1 + (size_t)ar * HID;
    const _Float16* bcol = w2T + l16 * HID;

    v8f acc = {};
#pragma unroll
    for (int k0 = 0; k0 < HID; k0 += 32) {
        const int ka = k0 + half * 16;
        const int kb = k0 + half * 16;
        v16h a;
#pragma unroll
        for (int i = 0; i < 8; ++i) {
            a[i]     = (_Float16)fmaxf(arow[ka + i],      0.f);
            a[i + 8] = (_Float16)fmaxf(arow[ka + 8 + i], 0.f);
        }
        v16h b = *(const v16h*)(bcol + kb);
        acc = __builtin_amdgcn_wmma_f32_16x16x32_f16(false, a, false, b,
                                                     (short)0, acc, false, false);
    }
    if (l16 < OUT_DIM) {                   // drop the zero-padded columns
        const int m0 = row0 + half * 8;
        float* outp = z + (size_t)m0 * OUT_DIM + l16;
        if (row0 + 16 <= n) {
#pragma unroll
            for (int r = 0; r < 8; ++r)
                outp[(size_t)r * OUT_DIM] = acc[r];
        } else {
#pragma unroll
            for (int r = 0; r < 8; ++r)
                if (m0 + r < n) outp[(size_t)r * OUT_DIM] = acc[r];
        }
    }
}

// ---------------------------------------------------------------------------
// GCN aggregation 2 + log_softmax (in-place on d_out)
// ---------------------------------------------------------------------------
__global__ void k_agg2_init(const float* __restrict__ z, const float* __restrict__ dinv,
                            const float* __restrict__ b2, float* __restrict__ out, int n) {
    int g = blockIdx.x * blockDim.x + threadIdx.x;
    if (g >= n * OUT_DIM) return;
    int nn = g / OUT_DIM, o = g - nn * OUT_DIM;
    float di = dinv[nn];
    out[g] = z[g] * di * di + b2[o];
}

__global__ void k_agg2_edges(const int* __restrict__ ei, const float* __restrict__ z,
                             const float* __restrict__ dinv, float* __restrict__ out, int e) {
    int g = blockIdx.x * blockDim.x + threadIdx.x;
    if (g >= e) return;
    int src = ei[g], dst = ei[e + g];
    float norm = dinv[src] * dinv[dst];
    const float* zs = z + (size_t)src * OUT_DIM;
    float* op = out + (size_t)dst * OUT_DIM;
#pragma unroll
    for (int o = 0; o < OUT_DIM; ++o) atomicAdd(op + o, zs[o] * norm);
}

__global__ void k_logsoftmax(float* out, int n) {
    int g = blockIdx.x * blockDim.x + threadIdx.x;
    if (g >= n) return;
    float* p = out + (size_t)g * OUT_DIM;
    float v[OUT_DIM];
    float m = p[0]; v[0] = p[0];
#pragma unroll
    for (int o = 1; o < OUT_DIM; ++o) { v[o] = p[o]; m = fmaxf(m, v[o]); }
    float s = 0.f;
#pragma unroll
    for (int o = 0; o < OUT_DIM; ++o) s += expf(v[o] - m);
    float ls = logf(s);
#pragma unroll
    for (int o = 0; o < OUT_DIM; ++o) p[o] = v[o] - m - ls;
}

// ---------------------------------------------------------------------------
extern "C" void kernel_launch(void* const* d_in, const int* in_sizes, int n_in,
                              void* d_out, int out_size, void* d_ws, size_t ws_size,
                              hipStream_t stream) {
    const float* x   = (const float*)d_in[0];
    const float* emb = (const float*)d_in[1];
    const float* W1  = (const float*)d_in[2];
    const float* b1  = (const float*)d_in[3];
    const float* W2  = (const float*)d_in[4];
    const float* b2  = (const float*)d_in[5];
    const int*   ei  = (const int*)d_in[6];
    float* out = (float*)d_out;

    const int N = in_sizes[0] / F_DIM;     // 20000
    const int E = in_sizes[6] / 2;         // 320000

    // workspace carve-up (~21.3 MB total); every byte rewritten each call
    char* ws = (char*)d_ws;
    auto alignup = [](size_t v) { return (v + 255) & ~(size_t)255; };
    size_t off = 0;
    float* colsum = (float*)(ws + off); off += F_DIM * 4;
    float* colss  = (float*)(ws + off); off += F_DIM * 4;
    float* meanp  = (float*)(ws + off); off += F_DIM * 4;
    float* istdp  = (float*)(ws + off); off += F_DIM * 4;
    float* cvec   = (float*)(ws + off); off += HID * 4;    off = alignup(off);
    float* dinv   = (float*)(ws + off); off += (size_t)N * 4; off = alignup(off);
    _Float16* w1vT = (_Float16*)(ws + off); off += (size_t)F_DIM * HID * 2;
    _Float16* w2T  = (_Float16*)(ws + off); off += 16 * HID * 2; off = alignup(off);
    float* hpre = (float*)(ws + off); off += (size_t)N * HID * 4;
    float* agg1 = (float*)(ws + off); off += (size_t)N * HID * 4;
    float* zbuf = (float*)(ws + off); off += (size_t)N * OUT_DIM * 4;

    const int B  = 256;
    const int gN = (N + B - 1) / B;

    k_init          <<<gN, B, 0, stream>>>(colsum, colss, cvec, dinv, N);
    k_colstats      <<<200, 256, 0, stream>>>(x, colsum, colss, N);
    k_finalize_stats<<<1, 256, 0, stream>>>(colsum, colss, meanp, istdp, N);
    k_deg           <<<(E + B - 1) / B, B, 0, stream>>>(ei, dinv, E);
    k_dinv          <<<gN, B, 0, stream>>>(dinv, N);
    k_fold_emb      <<<F_DIM, HID, 0, stream>>>(emb, W1, cvec);
    k_prep_w1vT     <<<F_DIM, HID, 0, stream>>>(W1, w1vT);
    k_prep_w2T      <<<16, HID, 0, stream>>>(W2, w2T);

    dim3 g1((N + 15) / 16, HID / 16);      // 1250 x 8 tiles, 1 wave each
    k_gemm1_wmma    <<<g1, 32, 0, stream>>>(x, meanp, istdp, w1vT, cvec, hpre, N);

    k_agg1_init     <<<((size_t)N * HID + B - 1) / B, B, 0, stream>>>(hpre, dinv, b1, agg1, N);
    k_agg1_edges    <<<((size_t)E * 32 + B - 1) / B, B, 0, stream>>>(ei, hpre, dinv, agg1, E);

    k_gemm2_wmma    <<<(N + 15) / 16, 32, 0, stream>>>(agg1, w2T, zbuf, N);

    k_agg2_init     <<<((size_t)N * OUT_DIM + B - 1) / B, B, 0, stream>>>(zbuf, dinv, b2, out, N);
    k_agg2_edges    <<<(E + B - 1) / B, B, 0, stream>>>(ei, zbuf, dinv, out, E);
    k_logsoftmax    <<<gN, B, 0, stream>>>(out, N);
}
